// MambaDecoderLayer_13657996001384
// MI455X (gfx1250) — compile-verified, matrix operand order checked by reading, not run on hardware
//
#include <hip/hip_runtime.h>
#include <hip/hip_bf16.h>

#define DMODEL 1024
#define DINNER 2048
#define DSTATE 16
#define DCONV 4
#define DTRANK 64
#define DFFN 4096
#define XDIM 96            // DT_RANK + 2*D_STATE
#define XZLD 4096          // 2*DINNER

typedef __attribute__((ext_vector_type(16))) __bf16 v16bf;
typedef __attribute__((ext_vector_type(8)))  __bf16 v8bf;
typedef __attribute__((ext_vector_type(4)))  __bf16 v4bf;
typedef __attribute__((ext_vector_type(8)))  float  v8f;

// ---------------------------------------------------------------------------
// bf16 fragment loader (16x32 A tile / dual-layout B tile), ISA 7.12.2:
//   lane<16 : j<8 -> K=j,   j>=8 -> K=j+8    (0..7, 16..23)
//   lane>=16: j<8 -> K=j+8, j>=8 -> K=j+16   (8..15, 24..31)
// Two 16B chunks -> two global_load_b128, no conversion VALU.
// ---------------------------------------------------------------------------
__device__ __forceinline__ v16bf load_frag(const __bf16* __restrict__ base,
                                           int row, int ld, int k0, int lane) {
  const __bf16* p = base + (size_t)row * (size_t)ld + k0 + ((lane >> 4) << 3);
  v8bf lo = *reinterpret_cast<const v8bf*>(p);
  v8bf hi = *reinterpret_cast<const v8bf*>(p + 16);
  return __builtin_shufflevector(lo, hi, 0, 1, 2, 3, 4, 5, 6, 7,
                                 8, 9, 10, 11, 12, 13, 14, 15);
}

// ---------------------------------------------------------------------------
// GEMM: C[M,N] = A[M,K] * W[N,K]^T   (bf16 in, f32 WMMA accumulate, f32 out)
// 8 waves/block as 2(m) x 4(n); wave tile 64x32 = 4x2 wmma frags.
// Block tile 128x128. M % 128 == 0, N % 16 == 0, K % 32 == 0.
// ---------------------------------------------------------------------------
__global__ __launch_bounds__(256)
void gemm_wmma_kernel(const __bf16* __restrict__ A, int lda,
                      const __bf16* __restrict__ W, int ldw,
                      float* __restrict__ C, int ldc,
                      int M, int N, int K) {
  const int lane = threadIdx.x & 31;
  const int wave = threadIdx.x >> 5;
  const int wm = wave >> 2;               // 0..1
  const int wn = wave & 3;                // 0..3
  const int m0 = blockIdx.y * 128 + wm * 64;
  const int n0 = blockIdx.x * 128 + wn * 32;

  v8f acc0[4] = {v8f{}, v8f{}, v8f{}, v8f{}};
  v8f acc1[4] = {v8f{}, v8f{}, v8f{}, v8f{}};
  const int r = lane & 15;
  const bool nv0 = (n0 < N);
  const bool nv1 = (n0 + 16 < N);
  const int bn0 = nv0 ? (n0 + r) : 0;       // clamp OOB col tiles (stores skipped)
  const int bn1 = nv1 ? (n0 + 16 + r) : 0;

  for (int k = 0; k < K; k += 32) {
    if (k + 32 < K) {
      __builtin_prefetch(A + (size_t)(m0 + r) * lda + k + 32, 0, 1);
      __builtin_prefetch(W + (size_t)bn0 * ldw + k + 32, 0, 1);
    }
    v16bf b0 = load_frag(W, bn0, ldw, k, lane);
    v16bf b1 = load_frag(W, bn1, ldw, k, lane);
#pragma unroll
    for (int i = 0; i < 4; ++i) {
      v16bf a = load_frag(A, m0 + 16 * i + r, lda, k, lane);
      acc0[i] = __builtin_amdgcn_wmma_f32_16x16x32_bf16(false, a, false, b0,
                                                        (short)0, acc0[i], false, false);
      acc1[i] = __builtin_amdgcn_wmma_f32_16x16x32_bf16(false, a, false, b1,
                                                        (short)0, acc1[i], false, false);
    }
  }

  // C/D layout: VGPR i, lanes 0-15 -> M=i, lanes 16-31 -> M=i+8; N = lane&15.
  const int rowBase = m0 + ((lane >> 4) << 3);
  const int col0 = n0 + (lane & 15);
  const int col1 = col0 + 16;
#pragma unroll
  for (int i = 0; i < 4; ++i) {
#pragma unroll
    for (int j = 0; j < 8; ++j) {
      int row = rowBase + 16 * i + j;
      if (nv0) C[(size_t)row * ldc + col0] = acc0[i][j];
      if (nv1) C[(size_t)row * ldc + col1] = acc1[i][j];
    }
  }
}

// f32 -> bf16 bulk convert (4 elements / thread)
__global__ void cvt_bf16_kernel(const float* __restrict__ in,
                                __bf16* __restrict__ out, long long n4) {
  long long i = (long long)blockIdx.x * blockDim.x + threadIdx.x;
  if (i >= n4) return;
  float4 v = reinterpret_cast<const float4*>(in)[i];
  v4bf o;
  o[0] = (__bf16)v.x; o[1] = (__bf16)v.y; o[2] = (__bf16)v.z; o[3] = (__bf16)v.w;
  reinterpret_cast<v4bf*>(out)[i] = o;
}

// xdbl[:, 0:64] (f32, ld 96) -> packed bf16 [M,64]
__global__ void cvt_dt_kernel(const float* __restrict__ xdbl,
                              __bf16* __restrict__ out, long long total) {
  long long i = (long long)blockIdx.x * blockDim.x + threadIdx.x;
  if (i >= total) return;
  long long rr = i / DTRANK;
  int c = (int)(i % DTRANK);
  out[i] = (__bf16)xdbl[rr * XDIM + c];
}

// causal depthwise conv1d (k=4) + bias + SiLU; writes f32 (scan) and bf16 (GEMM)
__global__ void conv_silu_kernel(const float* __restrict__ xz,
                                 const float* __restrict__ cw,
                                 const float* __restrict__ cb,
                                 float* __restrict__ xout,
                                 __bf16* __restrict__ xbf,
                                 int L, long long total) {
  long long i = (long long)blockIdx.x * blockDim.x + threadIdx.x;
  if (i >= total) return;
  int d = (int)(i % DINNER);
  long long bt = i / DINNER;     // b*L + t
  int t = (int)(bt % L);
  float acc = cb[d];
#pragma unroll
  for (int kk = 0; kk < DCONV; ++kk) {
    int dtp = kk - (DCONV - 1);          // -3..0
    if (t + dtp >= 0)
      acc += cw[d * DCONV + kk] * xz[(bt + dtp) * (long long)XZLD + d];
  }
  acc = acc / (1.f + __expf(-acc));       // SiLU
  xout[i] = acc;
  xbf[i] = (__bf16)acc;
}

// dt = softplus(dt_raw + dt_b)  (in place, f32 for scan)
__global__ void bias_softplus_kernel(float* __restrict__ dt,
                                     const float* __restrict__ db,
                                     long long total) {
  long long i = (long long)blockIdx.x * blockDim.x + threadIdx.x;
  if (i >= total) return;
  int d = (int)(i % DINNER);
  float v = dt[i] + db[d];
  dt[i] = (v > 20.f) ? v : log1pf(__expf(v));
}

// out_bf16 = relu(h + b)
__global__ void bias_relu_bf16_kernel(const float* __restrict__ h,
                                      const float* __restrict__ b,
                                      __bf16* __restrict__ out,
                                      long long total, int n) {
  long long i = (long long)blockIdx.x * blockDim.x + threadIdx.x;
  if (i >= total) return;
  float v = h[i] + b[i % n];
  out[i] = (__bf16)(v > 0.f ? v : 0.f);
}

// dst_bf16[b][0:L][:] = mem[b], dst_bf16[b][L:2L][:] = tgt[b]
__global__ void concat_bf16_kernel(const float* __restrict__ mem,
                                   const float* __restrict__ tgt,
                                   __bf16* __restrict__ dst,
                                   long long totalHalf, int L) {
  long long i = (long long)blockIdx.x * blockDim.x + threadIdx.x;
  if (i >= totalHalf) return;
  long long LD = (long long)L * DMODEL;
  long long b = i / LD;
  long long rc = i % LD;
  dst[b * 2 * LD + rc]      = (__bf16)mem[i];
  dst[b * 2 * LD + LD + rc] = (__bf16)tgt[i];
}

// ---------------------------------------------------------------------------
// Selective scan: thread = (batch, channel), h[16] in VGPRs, B_t/C_t via LDS.
// Emits y directly as bf16 (only consumed by the out-projection GEMM).
// ---------------------------------------------------------------------------
__global__ __launch_bounds__(256)
void scan_kernel(const float* __restrict__ x,    // [B,L,DINNER] conv output
                 const float* __restrict__ xz,   // [B,L,4096] (z at +DINNER)
                 const float* __restrict__ xdbl, // [B,L,96]
                 const float* __restrict__ dt,   // [B,L,DINNER]
                 const float* __restrict__ A_log,
                 const float* __restrict__ Dp,
                 __bf16* __restrict__ yout,      // [B,L,DINNER]
                 int L) {
  __shared__ float sB[DSTATE], sC[DSTATE];
  const int b = blockIdx.y;
  const int d = blockIdx.x * 256 + threadIdx.x;
  float a[DSTATE], h[DSTATE];
#pragma unroll
  for (int n = 0; n < DSTATE; ++n) {
    a[n] = -__expf(A_log[d * DSTATE + n]);
    h[n] = 0.f;
  }
  const float Dv = Dp[d];
  const long long rowz = (long long)b * L;
  for (int t = 0; t < L; ++t) {
    const long long row = rowz + t;
    __syncthreads();
    if (threadIdx.x < DSTATE)
      sB[threadIdx.x] = xdbl[row * XDIM + DTRANK + threadIdx.x];
    else if (threadIdx.x < 2 * DSTATE)
      sC[threadIdx.x - DSTATE] = xdbl[row * XDIM + DTRANK + threadIdx.x];
    __syncthreads();
    const float dtv = dt[row * DINNER + d];
    const float xv  = x[row * DINNER + d];
    const float zv  = xz[row * XZLD + DINNER + d];
    const float dtx = dtv * xv;
    float y = 0.f;
#pragma unroll
    for (int n = 0; n < DSTATE; ++n) {
      float e = __expf(dtv * a[n]);
      h[n] = e * h[n] + dtx * sB[n];
      y += h[n] * sC[n];
    }
    y += Dv * xv;
    const float sil = zv / (1.f + __expf(-zv));
    yout[row * DINNER + d] = (__bf16)(y * sil);
  }
}

// ---------------------------------------------------------------------------
// out[r] = LayerNorm( yg[map(r)] + resid[r] (+ bias) ) * g + b
// map(r): r = b*L + t -> yrow = b*Ly + off + t   (handles [:, -Lt:] slice)
// optional extra bf16 output copy.
// ---------------------------------------------------------------------------
__global__ __launch_bounds__(256)
void add_ln_kernel(const float* __restrict__ yg,
                   const float* __restrict__ resid,
                   const float* __restrict__ bias,   // nullable, len DMODEL
                   const float* __restrict__ g,
                   const float* __restrict__ bvec,
                   float* __restrict__ out,
                   __bf16* __restrict__ outb,        // nullable
                   int L, int Ly, int off) {
  const int r = blockIdx.x;
  const int b = r / L, t = r % L;
  const long long yrow = (long long)(b * Ly + off + t) * DMODEL;
  const long long orow = (long long)r * DMODEL;
  float v[4];
  float s = 0.f, s2 = 0.f;
#pragma unroll
  for (int j = 0; j < 4; ++j) {
    int c = threadIdx.x + j * 256;
    float xv = yg[yrow + c] + resid[orow + c];
    if (bias) xv += bias[c];
    v[j] = xv; s += xv; s2 += xv * xv;
  }
  __shared__ float red[256], red2[256];
  red[threadIdx.x] = s; red2[threadIdx.x] = s2;
  __syncthreads();
  for (int st = 128; st > 0; st >>= 1) {
    if (threadIdx.x < st) {
      red[threadIdx.x]  += red[threadIdx.x + st];
      red2[threadIdx.x] += red2[threadIdx.x + st];
    }
    __syncthreads();
  }
  const float mean = red[0] * (1.f / DMODEL);
  const float var  = red2[0] * (1.f / DMODEL) - mean * mean;
  const float inv  = rsqrtf(var + 1e-6f);
#pragma unroll
  for (int j = 0; j < 4; ++j) {
    int c = threadIdx.x + j * 256;
    float o = (v[j] - mean) * inv * g[c] + bvec[c];
    out[orow + c] = o;
    if (outb) outb[orow + c] = (__bf16)o;
  }
}

// ---------------------------------------------------------------------------
// Host-side helpers
// ---------------------------------------------------------------------------
static inline void launch_gemm(hipStream_t s,
                               const __bf16* A, int lda,
                               const __bf16* W, int ldw,
                               float* C, int ldc,
                               int M, int N, int K) {
  dim3 grid((N + 127) / 128, M / 128);
  gemm_wmma_kernel<<<grid, 256, 0, s>>>(A, lda, W, ldw, C, ldc, M, N, K);
}

static inline void launch_cvt(hipStream_t s, const float* in, __bf16* out, long long n) {
  long long n4 = n / 4;
  cvt_bf16_kernel<<<(unsigned)((n4 + 255) / 256), 256, 0, s>>>(in, out, n4);
}

static void run_mamba(hipStream_t s, const __bf16* ubf, int M /*rows=B*L*/, int L,
                      const __bf16* in_wb, const float* conv_w, const float* conv_b,
                      const __bf16* x_wb, const __bf16* dt_wb, const float* dt_b,
                      const float* A_log, const float* Dp, const __bf16* out_wb,
                      float* xz, float* xf32, __bf16* xbf, float* xdbl,
                      __bf16* dt64b, float* dtb, __bf16* ybf, float* mout) {
  const long long total = (long long)M * DINNER;
  const int B = M / L;
  // 1. xz = u @ in_w.T                        [M, 4096]
  launch_gemm(s, ubf, DMODEL, in_wb, DMODEL, xz, XZLD, M, 2 * DINNER, DMODEL);
  // 2. causal conv + SiLU -> xf32 + xbf       [M, 2048]
  conv_silu_kernel<<<(unsigned)((total + 255) / 256), 256, 0, s>>>(
      xz, conv_w, conv_b, xf32, xbf, L, total);
  // 3. x_dbl = x @ x_w.T                      [M, 96]
  launch_gemm(s, xbf, DINNER, x_wb, DINNER, xdbl, XDIM, M, XDIM, DINNER);
  // 3b. pack dt-rank columns to bf16          [M, 64]
  long long tdt = (long long)M * DTRANK;
  cvt_dt_kernel<<<(unsigned)((tdt + 255) / 256), 256, 0, s>>>(xdbl, dt64b, tdt);
  // 4. dt_raw = x_dbl[:, :64] @ dt_w.T        [M, 2048]
  launch_gemm(s, dt64b, DTRANK, dt_wb, DTRANK, dtb, DINNER, M, DINNER, DTRANK);
  // 5. dt = softplus(dt_raw + dt_b)
  bias_softplus_kernel<<<(unsigned)((total + 255) / 256), 256, 0, s>>>(dtb, dt_b, total);
  // 6. selective scan -> ybf (bf16)
  dim3 sg(DINNER / 256, B);
  scan_kernel<<<sg, 256, 0, s>>>(xf32, xz, xdbl, dtb, A_log, Dp, ybf, L);
  // 7. mamba_out = y @ out_w.T                [M, 1024]
  launch_gemm(s, ybf, DINNER, out_wb, DINNER, mout, DMODEL, M, DMODEL, DINNER);
}

extern "C" void kernel_launch(void* const* d_in, const int* in_sizes, int n_in,
                              void* d_out, int out_size, void* d_ws, size_t ws_size,
                              hipStream_t stream) {
  const float* tgt    = (const float*)d_in[0];
  const float* memory = (const float*)d_in[1];
  const float* sm[9]; const float* cm[9];
  for (int i = 0; i < 9; ++i) { sm[i] = (const float*)d_in[2 + i]; cm[i] = (const float*)d_in[11 + i]; }
  const float* ln1_g = (const float*)d_in[20];
  const float* ln1_b = (const float*)d_in[21];
  const float* ln2_g = (const float*)d_in[22];
  const float* ln2_b = (const float*)d_in[23];
  const float* ln3_g = (const float*)d_in[24];
  const float* ln3_b = (const float*)d_in[25];
  const float* ffn_w1 = (const float*)d_in[26];
  const float* ffn_b1 = (const float*)d_in[27];
  const float* ffn_w2 = (const float*)d_in[28];
  const float* ffn_b2 = (const float*)d_in[29];

  const int B = 4, Lt = 1024;
  const int Mself  = B * Lt;       // 4096
  const int Mcross = B * 2 * Lt;   // 8192

  // ---- workspace arena ----
  char* p = (char*)d_ws;
  auto alloc_f = [&](size_t n) { float* r = (float*)p;  p += n * sizeof(float);  return r; };
  auto alloc_b = [&](size_t n) { __bf16* r = (__bf16*)p; p += n * sizeof(__bf16); return r; };

  float*  xz    = alloc_f((size_t)Mcross * XZLD);     // 128MB (FFN hidden f32 reuses it)
  float*  xf32  = alloc_f((size_t)Mcross * DINNER);   // 64MB
  float*  xdbl  = alloc_f((size_t)Mcross * XDIM);
  float*  dtb   = alloc_f((size_t)Mcross * DINNER);   // 64MB
  float*  mout  = alloc_f((size_t)Mcross * DMODEL);   // 32MB
  float*  tgt1  = alloc_f((size_t)Mself * DMODEL);
  float*  tgt2  = alloc_f((size_t)Mself * DMODEL);
  __bf16* w_in_s  = alloc_b((size_t)2 * DINNER * DMODEL);
  __bf16* w_x_s   = alloc_b((size_t)XDIM * DINNER);
  __bf16* w_dt_s  = alloc_b((size_t)DINNER * DTRANK);
  __bf16* w_out_s = alloc_b((size_t)DMODEL * DINNER);
  __bf16* w_in_c  = alloc_b((size_t)2 * DINNER * DMODEL);
  __bf16* w_x_c   = alloc_b((size_t)XDIM * DINNER);
  __bf16* w_dt_c  = alloc_b((size_t)DINNER * DTRANK);
  __bf16* w_out_c = alloc_b((size_t)DMODEL * DINNER);
  __bf16* w_f1    = alloc_b((size_t)DFFN * DMODEL);
  __bf16* w_f2    = alloc_b((size_t)DMODEL * DFFN);
  __bf16* tgtbf   = alloc_b((size_t)Mself * DMODEL);
  __bf16* cinbf   = alloc_b((size_t)Mcross * DMODEL);
  __bf16* xbf     = alloc_b((size_t)Mcross * DINNER);
  __bf16* dt64b   = alloc_b((size_t)Mcross * DTRANK);
  __bf16* ybf     = alloc_b((size_t)Mcross * DINNER);
  __bf16* tgt2bf  = alloc_b((size_t)Mself * DMODEL);
  __bf16* ffnhbf  = alloc_b((size_t)Mself * DFFN);
  float*  ffnh    = xz;   // [4096,4096] f32, xz region is dead during FFN

  // ---- weight / input bf16 conversion (bandwidth-trivial) ----
  launch_cvt(stream, sm[0], w_in_s,  (long long)2 * DINNER * DMODEL);
  launch_cvt(stream, sm[3], w_x_s,   (long long)XDIM * DINNER);
  launch_cvt(stream, sm[4], w_dt_s,  (long long)DINNER * DTRANK);
  launch_cvt(stream, sm[8], w_out_s, (long long)DMODEL * DINNER);
  launch_cvt(stream, cm[0], w_in_c,  (long long)2 * DINNER * DMODEL);
  launch_cvt(stream, cm[3], w_x_c,   (long long)XDIM * DINNER);
  launch_cvt(stream, cm[4], w_dt_c,  (long long)DINNER * DTRANK);
  launch_cvt(stream, cm[8], w_out_c, (long long)DMODEL * DINNER);
  launch_cvt(stream, ffn_w1, w_f1,   (long long)DFFN * DMODEL);
  launch_cvt(stream, ffn_w2, w_f2,   (long long)DMODEL * DFFN);
  launch_cvt(stream, tgt, tgtbf,     (long long)Mself * DMODEL);

  // ---- self mamba + LN1 ----
  run_mamba(stream, tgtbf, Mself, Lt, w_in_s, sm[1], sm[2], w_x_s, w_dt_s, sm[5],
            sm[6], sm[7], w_out_s, xz, xf32, xbf, xdbl, dt64b, dtb, ybf, mout);
  add_ln_kernel<<<Mself, 256, 0, stream>>>(mout, tgt, nullptr, ln1_g, ln1_b,
                                           tgt1, nullptr, Lt, Lt, 0);

  // ---- cross mamba on concat(memory, tgt1) + LN2 (take last Lt rows) ----
  long long halfTot = (long long)Mself * DMODEL;
  concat_bf16_kernel<<<(unsigned)((halfTot + 255) / 256), 256, 0, stream>>>(
      memory, tgt1, cinbf, halfTot, Lt);
  run_mamba(stream, cinbf, Mcross, 2 * Lt, w_in_c, cm[1], cm[2], w_x_c, w_dt_c, cm[5],
            cm[6], cm[7], w_out_c, xz, xf32, xbf, xdbl, dt64b, dtb, ybf, mout);
  add_ln_kernel<<<Mself, 256, 0, stream>>>(mout, tgt1, nullptr, ln2_g, ln2_b,
                                           tgt2, tgt2bf, Lt, 2 * Lt, Lt);

  // ---- FFN + LN3 -> d_out ----
  launch_gemm(stream, tgt2bf, DMODEL, w_f1, DMODEL, ffnh, DFFN, Mself, DFFN, DMODEL);
  long long ffnTot = (long long)Mself * DFFN;
  bias_relu_bf16_kernel<<<(unsigned)((ffnTot + 255) / 256), 256, 0, stream>>>(
      ffnh, ffn_b1, ffnhbf, ffnTot, DFFN);
  launch_gemm(stream, ffnhbf, DFFN, w_f2, DFFN, mout, DMODEL, Mself, DMODEL, DFFN);
  add_ln_kernel<<<Mself, 256, 0, stream>>>(mout, tgt2, ffn_b2, ln3_g, ln3_b,
                                           (float*)d_out, nullptr, Lt, Lt, 0);
}
